// TFBNC_31301721653424
// MI455X (gfx1250) — matrix operand order repaired and checked by reading, not compiled, not run on hardware
//
#include <hip/hip_runtime.h>
#include <hip/hip_bf16.h>
#include <math.h>

typedef __bf16 bf16_t;
typedef bf16_t bf16x8 __attribute__((ext_vector_type(8)));
typedef bf16_t v16bf  __attribute__((ext_vector_type(16)));
typedef float  v8f    __attribute__((ext_vector_type(8)));

#define B_      128
#define N_      200
#define T_      360
#define P_      400
#define NPAD    224          // N padded to multiple of 32 (K dim)
#define STRIDE_ 25
#define NWIN    14           // T_ / STRIDE_
#define EPS_    1e-9f

// ---------------- workspace layout (bytes, 256-aligned regions) ----------------
#define OFF_NPAT 0u                       // bf16 [P_][NPAD]          179200
#define OFF_TIDX 179200u                  // int  [P_][3]               4800
#define OFF_TVAL 184064u                  // f32  [P_][3]               4800
#define OFF_MG   188928u                  // f32  [N_*N_]             160000
#define OFF_NACC 348928u                  // f32  [1] (pad 256)
#define OFF_COEF 349184u                  // f32  [P_][2]               3200
#define OFF_FS   352512u                  // f32  [B_][P_]            204800
#define WS_END   557312u
#define ZERO_FLOATS ((WS_END - OFF_MG) / 4u)   // zero mg..fs each launch

__device__ __forceinline__ float wave_sum(float v) {
#pragma unroll
  for (int off = 16; off >= 1; off >>= 1) v += __shfl_xor(v, off, 32);
  return v;
}

__global__ void k_zero(float* __restrict__ p, int n) {
  int i = blockIdx.x * blockDim.x + threadIdx.x;
  if (i < n) p[i] = 0.0f;
}

// One wave per pattern row: normalize, emit bf16 npat (padded), top-3 |.|,
// maskgraph atomics, Critical_ROI norm accumulation.
__global__ void k_patterns(const float* __restrict__ patterns,
                           bf16_t* __restrict__ npat,
                           int* __restrict__ tidx, float* __restrict__ tval,
                           float* __restrict__ mg, float* __restrict__ nacc) {
  int wave = threadIdx.x >> 5;
  int lane = threadIdx.x & 31;
  int p = blockIdx.x * 4 + wave;
  if (p >= P_) return;
  const float* row = patterns + p * N_;

  float ss = 0.0f;
  for (int n = lane; n < N_; n += 32) { float v = row[n]; ss += v * v; }
  ss = wave_sum(ss);
  float pinv = 1.0f / sqrtf(ss + EPS_);

  for (int n = lane; n < NPAD; n += 32)
    npat[p * NPAD + n] = (bf16_t)((n < N_) ? row[n] * pinv : 0.0f);

  int   si[3];
  float sv[3];
#pragma unroll
  for (int k = 0; k < 3; ++k) {
    float bm = -1.0f;
    int   bi = 0x7fffffff;
    for (int n = lane; n < N_; n += 32) {
      bool used = false;
      for (int j = 0; j < k; ++j) used |= (si[j] == n);
      if (used) continue;
      float a = fabsf(row[n]) * pinv;
      if (a > bm || (a == bm && n < bi)) { bm = a; bi = n; }
    }
#pragma unroll
    for (int off = 16; off >= 1; off >>= 1) {
      float ov = __shfl_xor(bm, off, 32);
      int   oi = __shfl_xor(bi, off, 32);
      if (ov > bm || (ov == bm && oi < bi)) { bm = ov; bi = oi; }
    }
    si[k] = bi;
    sv[k] = row[bi] * pinv;   // signed normalized value (realpattern entry)
  }

  if (lane == 0) {
#pragma unroll
    for (int k = 0; k < 3; ++k) { tidx[p * 3 + k] = si[k]; tval[p * 3 + k] = sv[k]; }
#pragma unroll
    for (int i = 0; i < 3; ++i)
#pragma unroll
      for (int j = 0; j < 3; ++j)
        atomicAdd(&mg[si[i] * N_ + si[j]], 1.0f);
    float tn = sqrtf(sv[0] * sv[0] + sv[1] * sv[1] + sv[2] * sv[2]);
    float pn = sqrtf(ss) * pinv;
    float d  = 1.0f - tn / pn;
    atomicAdd(nacc, d * d);
  }
}

// Fold graphs->W_cls through the 3x3 sparse outer product per pattern:
// coef[p][c] = sum_{i,j} v_i v_j / (mg + eps) * W[(si*N+sj)*2 + c]
__global__ void k_coef(const int* __restrict__ tidx, const float* __restrict__ tval,
                       const float* __restrict__ mg, const float* __restrict__ W,
                       float* __restrict__ coef) {
  int p = blockIdx.x * blockDim.x + threadIdx.x;
  if (p >= P_) return;
  int si[3]; float sv[3];
#pragma unroll
  for (int k = 0; k < 3; ++k) { si[k] = tidx[p * 3 + k]; sv[k] = tval[p * 3 + k]; }
  float c0 = 0.0f, c1 = 0.0f;
#pragma unroll
  for (int i = 0; i < 3; ++i)
#pragma unroll
    for (int j = 0; j < 3; ++j) {
      int   id  = si[i] * N_ + si[j];
      float wgt = sv[i] * sv[j] / (mg[id] + EPS_);
      c0 += wgt * W[id * 2 + 0];
      c1 += wgt * W[id * 2 + 1];
    }
  coef[p * 2 + 0] = c0;
  coef[p * 2 + 1] = c1;
}

// One block per (window w, batch b): build normalized-frame bf16 tile in LDS,
// WMMA against all patterns, in-register window max, atomic add max/sap.
__global__ __launch_bounds__(256) void k_window(const float* __restrict__ x,
                                                const int* __restrict__ length,
                                                const bf16_t* __restrict__ npat,
                                                float* __restrict__ fs) {
  __shared__ __attribute__((aligned(16))) bf16_t Asm[32 * NPAD];
  __shared__ float psum[8 * 32];
  __shared__ float inv_s[32];

  int b = blockIdx.y, w = blockIdx.x;
  int sap = length[b] / STRIDE_;
  if (w >= sap) return;   // fully masked window -> contributes exactly 0

  int tid  = threadIdx.x;
  int lane = tid & 31, wave = tid >> 5;
  int t0   = w * STRIDE_;

  // phase 1: per-frame sum of squares (lanes stride over t -> coalesced in T)
  {
    int   chunk = wave;          // 8 chunks x 25 nodes
    int   tl    = lane;          // 0..31 (t0+31 <= 356 < T_, safe)
    int   nbase = chunk * 25;
    float ss    = 0.0f;
    const float* xb = x + (b * N_) * T_ + t0 + tl;
    for (int k = 0; k < 25; ++k) {
      float v = xb[(nbase + k) * T_];
      ss += v * v;
    }
    psum[chunk * 32 + tl] = ss;
  }
  __syncthreads();
  if (tid < 32) {
    float s = 0.0f;
    for (int c = 0; c < 8; ++c) s += psum[c * 32 + tid];
    inv_s[tid] = (tid < 25) ? (1.0f / sqrtf(s + EPS_)) : 0.0f;
  }
  __syncthreads();

  // phase 2: normalized frames -> bf16 A tile [32 x NPAD], zero-padded
  for (int idx = tid; idx < 32 * NPAD; idx += 256) {
    int   tl = idx & 31;
    int   n  = idx >> 5;
    float v  = 0.0f;
    if (tl < 25 && n < N_)
      v = x[(b * N_ + n) * T_ + t0 + tl] * inv_s[tl];
    Asm[tl * NPAD + n] = (bf16_t)v;
  }
  __syncthreads();

  // phase 3: WMMA — each wave owns whole P-tiles (both M-tiles of the window)
  float inv_sap = 1.0f / (float)sap;
  int m = lane & 15, hi = lane >> 4;
  for (int pt = wave; pt < P_ / 16; pt += 8) {
    v8f acc0 = {0, 0, 0, 0, 0, 0, 0, 0};
    v8f acc1 = {0, 0, 0, 0, 0, 0, 0, 0};
    int prow = pt * 16 + m;
#pragma unroll
    for (int kk = 0; kk < NPAD / 32; ++kk) {
      // B fragment: lane = column (pattern), K = 8*hi + (e<8 ? e : e+8) -> chunks at +0,+16
      const bf16_t* bptr = npat + prow * NPAD + kk * 32 + hi * 8;
      bf16x8 b0 = *(const bf16x8*)(bptr);
      bf16x8 b1 = *(const bf16x8*)(bptr + 16);
      v16bf bfrag = __builtin_shufflevector(b0, b1, 0, 1, 2, 3, 4, 5, 6, 7,
                                            8, 9, 10, 11, 12, 13, 14, 15);
      // A fragment: lane = row, K = 8*hi + (e<8 ? e : e+8) -> chunks at +0,+16
      const bf16_t* a0p = &Asm[m * NPAD + kk * 32 + hi * 8];
      bf16x8 a0lo = *(const bf16x8*)(a0p);
      bf16x8 a0hi = *(const bf16x8*)(a0p + 16);
      v16bf afrag0 = __builtin_shufflevector(a0lo, a0hi, 0, 1, 2, 3, 4, 5, 6, 7,
                                             8, 9, 10, 11, 12, 13, 14, 15);
      const bf16_t* a1p = &Asm[(16 + m) * NPAD + kk * 32 + hi * 8];
      bf16x8 a1lo = *(const bf16x8*)(a1p);
      bf16x8 a1hi = *(const bf16x8*)(a1p + 16);
      v16bf afrag1 = __builtin_shufflevector(a1lo, a1hi, 0, 1, 2, 3, 4, 5, 6, 7,
                                             8, 9, 10, 11, 12, 13, 14, 15);
      acc0 = __builtin_amdgcn_wmma_f32_16x16x32_bf16(false, afrag0, false, bfrag,
                                                     (short)0, acc0, false, false);
      acc1 = __builtin_amdgcn_wmma_f32_16x16x32_bf16(false, afrag1, false, bfrag,
                                                     (short)0, acc1, false, false);
    }
    // strided max-pool over t = 0..24 straight from the D layout:
    // lane<16 holds rows {0..7} of each tile; lane>=16 rows {8..15}
    float mx = -3.4e38f;
#pragma unroll
    for (int r = 0; r < 8; ++r) mx = fmaxf(mx, acc0[r]);   // t 0..15 split by hi
    if (hi == 0) {
#pragma unroll
      for (int r = 0; r < 8; ++r) mx = fmaxf(mx, acc1[r]); // t 16..23
    } else {
      mx = fmaxf(mx, acc1[0]);                             // t 24
    }
    mx = fmaxf(mx, __shfl_xor(mx, 16, 32));                // combine row halves
    if (lane < 16)
      atomicAdd(&fs[b * P_ + pt * 16 + lane], mx * inv_sap);
  }
}

// out[b,:] = finalscore[b,:] @ coef + b_cls ; norm ; rec
__global__ void k_final(const float* __restrict__ fs, const float* __restrict__ coef,
                        const float* __restrict__ bc, const float* __restrict__ nacc,
                        float* __restrict__ out) {
  int b = threadIdx.x;
  if (b < B_) {
    float o0 = bc[0], o1 = bc[1];
    for (int p = 0; p < P_; ++p) {
      float f = fs[b * P_ + p];
      o0 += f * coef[2 * p + 0];
      o1 += f * coef[2 * p + 1];
    }
    out[b * 2 + 0] = o0;
    out[b * 2 + 1] = o1;
  }
  if (b == 0) {
    out[2 * B_ + 0] = nacc[0] / 800.0f;  // Critical_ROI norm
    out[2 * B_ + 1] = 0.0f;              // rec
  }
}

extern "C" void kernel_launch(void* const* d_in, const int* in_sizes, int n_in,
                              void* d_out, int out_size, void* d_ws, size_t ws_size,
                              hipStream_t stream) {
  (void)in_sizes; (void)n_in; (void)out_size; (void)ws_size;
  const float* x        = (const float*)d_in[0];
  const int*   length   = (const int*)d_in[1];
  const float* patterns = (const float*)d_in[2];
  const float* W_cls    = (const float*)d_in[3];
  const float* b_cls    = (const float*)d_in[4];
  float* out = (float*)d_out;

  char*   ws   = (char*)d_ws;
  bf16_t* npat = (bf16_t*)(ws + OFF_NPAT);
  int*    tidx = (int*)(ws + OFF_TIDX);
  float*  tval = (float*)(ws + OFF_TVAL);
  float*  mg   = (float*)(ws + OFF_MG);
  float*  nacc = (float*)(ws + OFF_NACC);
  float*  coef = (float*)(ws + OFF_COEF);
  float*  fs   = (float*)(ws + OFF_FS);

  k_zero<<<(ZERO_FLOATS + 255) / 256, 256, 0, stream>>>(mg, (int)ZERO_FLOATS);
  k_patterns<<<P_ / 4, 128, 0, stream>>>(patterns, npat, tidx, tval, mg, nacc);
  k_coef<<<(P_ + 255) / 256, 256, 0, stream>>>(tidx, tval, mg, W_cls, coef);
  dim3 grid(NWIN, B_);
  k_window<<<grid, 256, 0, stream>>>(x, length, npat, fs);
  k_final<<<1, 128, 0, stream>>>(fs, coef, b_cls, nacc, out);
}
